// ARModel_60447369723927
// MI455X (gfx1250) — matrix operand order repaired and checked by reading, not compiled, not run on hardware
//
#include <hip/hip_runtime.h>

// ---------------------------------------------------------------------------
// GRU seq2seq (encoder S=1024 steps, autoregressive decoder T=256 steps)
// B=512, I=64, H=512.  Persistent-kernel design for MI455X (gfx1250):
//   * 64 persistent workgroups = 8 row-blocks x 8 col-blocks of the [B,H] state
//   * per-WG weight slices converted to bf16 ONCE and kept in LDS (~287 KB)
//   * per-step GEMMs via v_wmma_f32_16x16x32_bf16 (f32 accumulate)
//   * grid-wide split barrier (global atomic + s_sleep spin) between steps
// ---------------------------------------------------------------------------

typedef __attribute__((ext_vector_type(16))) __bf16      v16bf;
typedef __attribute__((ext_vector_type(8)))  float        v8f;
typedef __attribute__((ext_vector_type(4)))  unsigned int v4u;
typedef __attribute__((ext_vector_type(8)))  unsigned int v8u;

namespace {
constexpr int Bb = 512;   // batch
constexpr int Ss = 1024;  // source sequence length
constexpr int Ii = 64;    // input / output feature dim
constexpr int Hh = 512;   // hidden dim
constexpr int Tt = 256;   // decode steps
constexpr int NWG = 64;   // 8 row-blocks x 8 col-blocks

// LDS layout (bytes). Rows padded by 16B to dodge bank conflicts on the
// 1-KB-strided per-lane B reads.
constexpr unsigned WH_PITCH = Hh * 2 + 16;                 // 1040
constexpr unsigned WI_PITCH = Ii * 2 + 16;                 // 144
constexpr unsigned WL_PITCH = Hh * 2 + 16;                 // 1040
constexpr unsigned WH_BASE  = 0;                           // 192 rows (3 gates x 64 units)
constexpr unsigned WI_BASE  = WH_BASE + 192u * WH_PITCH;   // 192 rows
constexpr unsigned WL_BASE  = WI_BASE + 192u * WI_PITCH;   // 64 rows
constexpr unsigned LDS_TOTAL = WL_BASE + 64u * WL_PITCH;   // 293,888 B < 320 KB
}

// ---- WMMA wrapper ---------------------------------------------------------
__device__ __forceinline__ v8f wmma_bf16(v16bf a, v16bf b, v8f c) {
  // D = A(16x32 bf16) * B(32x16 bf16) + C(16x16 f32)
  return __builtin_amdgcn_wmma_f32_16x16x32_bf16(
      /*neg_a=*/false, a, /*neg_b=*/false, b,
      /*c_mod=*/(short)0, c, /*reuse_a=*/false, /*reuse_b=*/false);
}

// ---- A operand: 16x32 bf16 tile, lane m=lane&15; lanes<16 hold K k0..k0+7 &
// k0+16..k0+23 (k0=0), lanes>=16 the same with k0=8 (per ISA 16-bit A layout).
__device__ __forceinline__ v16bf load_a_bf16(const __bf16* __restrict__ rowptr,
                                             int kbase, int lane) {
  const int k0 = kbase + ((lane & 16) ? 8 : 0);
  v4u lo = *(const v4u*)(rowptr + k0);       // K = k0 .. k0+7
  v4u hi = *(const v4u*)(rowptr + k0 + 16);  // K = k0+16 .. k0+23
  return __builtin_bit_cast(v16bf,
      __builtin_shufflevector(lo, hi, 0, 1, 2, 3, 4, 5, 6, 7));
}

// Same A layout but source is f32 (x tile) -> convert on the fly.
__device__ __forceinline__ v16bf load_a_f32(const float* __restrict__ rowptr,
                                            int kbase, int lane) {
  const int k0 = kbase + ((lane & 16) ? 8 : 0);
  const float4 f0 = *(const float4*)(rowptr + k0);
  const float4 f1 = *(const float4*)(rowptr + k0 + 4);
  const float4 f2 = *(const float4*)(rowptr + k0 + 16);
  const float4 f3 = *(const float4*)(rowptr + k0 + 20);
  v16bf a;
  a[0]  = (__bf16)f0.x; a[1]  = (__bf16)f0.y; a[2]  = (__bf16)f0.z; a[3]  = (__bf16)f0.w;
  a[4]  = (__bf16)f1.x; a[5]  = (__bf16)f1.y; a[6]  = (__bf16)f1.z; a[7]  = (__bf16)f1.w;
  a[8]  = (__bf16)f2.x; a[9]  = (__bf16)f2.y; a[10] = (__bf16)f2.z; a[11] = (__bf16)f2.w;
  a[12] = (__bf16)f3.x; a[13] = (__bf16)f3.y; a[14] = (__bf16)f3.z; a[15] = (__bf16)f3.w;
  return a;
}

// ---- B operand: 32x16 bf16, column n = lane&15 (= one row of the stored
// weight slice); lanes<16 hold K k0..k0+7 & k0+16..k0+23 (k0=kbase), lanes>=16 k0=kbase+8.
__device__ __forceinline__ v16bf load_b_lds(const char* smem, unsigned rowoff,
                                            int kbase, int lane) {
  const unsigned off = rowoff + (unsigned)(kbase + ((lane & 16) ? 8 : 0)) * 2u;
  v4u lo = *(const v4u*)(smem + off);
  v4u hi = *(const v4u*)(smem + off + 32);
  return __builtin_bit_cast(v16bf,
      __builtin_shufflevector(lo, hi, 0, 1, 2, 3, 4, 5, 6, 7));
}

// ---- activations (overflow-safe) ------------------------------------------
__device__ __forceinline__ float sigm_f(float x) {
  return 1.0f / (1.0f + __expf(-x));
}
__device__ __forceinline__ float tanh_f(float x) {
  const float e = __expf(-2.0f * fabsf(x));     // e in (0, 1]
  return copysignf((1.0f - e) / (1.0f + e), x);
}

// ---- grid-wide barrier ----------------------------------------------------
__device__ __forceinline__ void grid_sync(unsigned* cnt, unsigned target) {
  __threadfence();
  __syncthreads();
  if (threadIdx.x == 0) {
    __hip_atomic_fetch_add(cnt, 1u, __ATOMIC_ACQ_REL, __HIP_MEMORY_SCOPE_AGENT);
    while (__hip_atomic_load(cnt, __ATOMIC_ACQUIRE, __HIP_MEMORY_SCOPE_AGENT) < target)
      __builtin_amdgcn_s_sleep(2);
  }
  __syncthreads();
}

// ---- one GRU step: this WG computes its 64x64 tile of h_next --------------
template <bool ENC>
__device__ __forceinline__ void gru_step(
    const float* __restrict__ x, int t, const __bf16* __restrict__ ybuf,
    const __bf16* __restrict__ hcur, __bf16* __restrict__ hnxt,
    float* __restrict__ hf32,
    const float* __restrict__ bi, const float* __restrict__ bh,
    char* smem, int rb, int cb, int wave, int lane) {
  // acc[ns][0]=r (i+h), acc[ns][1]=z (i+h), acc[ns][2]=n input part,
  // acc[ns][3]=n hidden part
  v8f acc[2][4];
#pragma unroll
  for (int i = 0; i < 2; ++i)
#pragma unroll
    for (int j = 0; j < 4; ++j)
      acc[i][j] = (v8f){0.f, 0.f, 0.f, 0.f, 0.f, 0.f, 0.f, 0.f};

  const int msub  = wave >> 1;          // 0..3 : 16-row subtile
  const int nsub0 = (wave & 1) * 2;     // 0 or 2: first of two 16-col subtiles
  const int mrow  = rb * 64 + msub * 16 + (lane & 15);
  const __bf16* hrow = hcur + (size_t)mrow * Hh;

  // --- hidden contribution: K = 512 ---------------------------------------
  for (int ks = 0; ks < 16; ++ks) {
    const v16bf a = load_a_bf16(hrow, ks * 32, lane);
#pragma unroll
    for (int ns = 0; ns < 2; ++ns) {
      const unsigned c_local = (unsigned)((nsub0 + ns) * 16 + (lane & 15));
#pragma unroll
      for (int g = 0; g < 3; ++g) {
        const v16bf b =
            load_b_lds(smem, WH_BASE + (g * 64u + c_local) * WH_PITCH, ks * 32, lane);
        const int ai = (g == 2) ? 3 : g;
        acc[ns][ai] = wmma_bf16(a, b, acc[ns][ai]);
      }
    }
  }

  // --- input contribution: K = 64 ------------------------------------------
  const float*  xrow = ENC ? (x + ((size_t)mrow * Ss + t) * Ii) : nullptr;
  const __bf16* yrow = ENC ? nullptr : (ybuf + (size_t)mrow * Ii);
  if (ENC && (t + 1 < Ss)) __builtin_prefetch(xrow + Ii, 0, 1);  // next timestep
#pragma unroll
  for (int ks = 0; ks < 2; ++ks) {
    const v16bf a = ENC ? load_a_f32(xrow, ks * 32, lane)
                        : load_a_bf16(yrow, ks * 32, lane);
#pragma unroll
    for (int ns = 0; ns < 2; ++ns) {
      const unsigned c_local = (unsigned)((nsub0 + ns) * 16 + (lane & 15));
#pragma unroll
      for (int g = 0; g < 3; ++g) {
        const v16bf b =
            load_b_lds(smem, WI_BASE + (g * 64u + c_local) * WI_PITCH, ks * 32, lane);
        const int ai = (g == 2) ? 2 : g;
        acc[ns][ai] = wmma_bf16(a, b, acc[ns][ai]);
      }
    }
  }

  // --- elementwise GRU update ----------------------------------------------
#pragma unroll
  for (int ns = 0; ns < 2; ++ns) {
    const int n = cb * 64 + (nsub0 + ns) * 16 + (lane & 15);
    const float bir = bi[n],          bhr = bh[n];
    const float biz = bi[Hh + n],     bhz = bh[Hh + n];
    const float bin = bi[2 * Hh + n], bhn = bh[2 * Hh + n];
#pragma unroll
    for (int j = 0; j < 8; ++j) {
      const int m = rb * 64 + msub * 16 + j + ((lane & 16) ? 8 : 0);
      const size_t idx = (size_t)m * Hh + n;
      const float hp = hf32[idx];
      const float r  = sigm_f(acc[ns][0][j] + bir + bhr);
      const float z  = sigm_f(acc[ns][1][j] + biz + bhz);
      const float nn = tanh_f(acc[ns][2][j] + bin + r * (acc[ns][3][j] + bhn));
      const float hv = (1.0f - z) * nn + z * hp;
      hf32[idx] = hv;
      hnxt[idx] = (__bf16)hv;
    }
  }
}

// ---- output projection: y = h @ Wl.T + bl  (done by col-block-0 WGs) ------
__device__ __forceinline__ void linear_step(
    const __bf16* __restrict__ hcur, __bf16* __restrict__ ybuf,
    float* __restrict__ out, int t, const float* __restrict__ bl,
    char* smem, int rb, int wave, int lane) {
  v8f acc[2];
#pragma unroll
  for (int i = 0; i < 2; ++i)
    acc[i] = (v8f){0.f, 0.f, 0.f, 0.f, 0.f, 0.f, 0.f, 0.f};

  const int msub  = wave >> 1;
  const int nsub0 = (wave & 1) * 2;
  const int mrow  = rb * 64 + msub * 16 + (lane & 15);
  const __bf16* hrow = hcur + (size_t)mrow * Hh;

  for (int ks = 0; ks < 16; ++ks) {
    const v16bf a = load_a_bf16(hrow, ks * 32, lane);
#pragma unroll
    for (int ns = 0; ns < 2; ++ns) {
      const unsigned c_local = (unsigned)((nsub0 + ns) * 16 + (lane & 15));
      const v16bf b = load_b_lds(smem, WL_BASE + c_local * WL_PITCH, ks * 32, lane);
      acc[ns] = wmma_bf16(a, b, acc[ns]);
    }
  }
#pragma unroll
  for (int ns = 0; ns < 2; ++ns) {
    const int n  = (nsub0 + ns) * 16 + (lane & 15);
    const float b0 = bl[n];
#pragma unroll
    for (int j = 0; j < 8; ++j) {
      const int m = rb * 64 + msub * 16 + j + ((lane & 16) ? 8 : 0);
      const float y = acc[ns][j] + b0;
      ybuf[(size_t)m * Ii + n] = (__bf16)y;           // next decoder input
      if (t >= 0) out[((size_t)m * Tt + t) * Ii + n] = y;
    }
  }
}

// ---------------------------------------------------------------------------
extern "C" __global__ void __launch_bounds__(256, 1)
gru_seq2seq_persistent(const float* __restrict__ x,   // [B,S,I]
                       const float* __restrict__ Wi,  // [3H,I]
                       const float* __restrict__ Wh,  // [3H,H]
                       const float* __restrict__ bi,  // [3H]
                       const float* __restrict__ bh,  // [3H]
                       const float* __restrict__ Wl,  // [I,H]
                       const float* __restrict__ bl,  // [I]
                       float* __restrict__ out,       // [B,T,I]
                       unsigned* __restrict__ cnt,    // grid barrier counter
                       float* __restrict__ hf32,      // [B,H] f32 state
                       __bf16* __restrict__ hb0,      // [B,H] bf16 state (ping)
                       __bf16* __restrict__ hb1,      // [B,H] bf16 state (pong)
                       __bf16* __restrict__ ybuf) {   // [B,I] decoder input
  extern __shared__ char smem[];
  const int wg   = blockIdx.x;       // 0..63
  const int rb   = wg >> 3;          // row block  (64 rows of B)
  const int cb   = wg & 7;           // col block  (64 cols of H)
  const int tid  = threadIdx.x;
  const int wave = tid >> 5;
  const int lane = tid & 31;

  // ---- one-time weight staging: f32 -> bf16 into LDS ----------------------
  // Wh slice: rows (g*64 + c) = gate g, hidden unit cb*64+c ; 512 K values.
  for (int idx = tid; idx < 192 * 512; idx += 256) {
    const int r = idx >> 9, k = idx & 511;
    const int g = r >> 6,   c = r & 63;
    const float v = Wh[(size_t)(g * Hh + cb * 64 + c) * Hh + k];
    *(__bf16*)(smem + WH_BASE + (unsigned)r * WH_PITCH + (unsigned)k * 2) = (__bf16)v;
  }
  // Wi slice: 192 rows x 64 K
  for (int idx = tid; idx < 192 * 64; idx += 256) {
    const int r = idx >> 6, k = idx & 63;
    const int g = r >> 6,   c = r & 63;
    const float v = Wi[(size_t)(g * Hh + cb * 64 + c) * Ii + k];
    *(__bf16*)(smem + WI_BASE + (unsigned)r * WI_PITCH + (unsigned)k * 2) = (__bf16)v;
  }
  // Wl: 64 rows x 512 K (used by cb==0 WGs; staged everywhere for simplicity)
  for (int idx = tid; idx < 64 * 512; idx += 256) {
    const int r = idx >> 9, k = idx & 511;
    const float v = Wl[(size_t)r * Hh + k];
    *(__bf16*)(smem + WL_BASE + (unsigned)r * WL_PITCH + (unsigned)k * 2) = (__bf16)v;
  }
  __syncthreads();

  unsigned ep = 0;
  __bf16* hcur = hb0;   // zeroed by host-side memset (h0 = 0)
  __bf16* hnxt = hb1;

  // ---- encoder: S steps ---------------------------------------------------
  for (int t = 0; t < Ss; ++t) {
    gru_step<true>(x, t, nullptr, hcur, hnxt, hf32, bi, bh, smem, rb, cb, wave, lane);
    grid_sync(cnt, (++ep) * NWG);
    __bf16* tmp = hcur; hcur = hnxt; hnxt = tmp;
  }

  // ---- y0 = h_n @ Wl.T + bl (decoder seed, not an output) -----------------
  if (cb == 0) linear_step(hcur, ybuf, out, -1, bl, smem, rb, wave, lane);
  grid_sync(cnt, (++ep) * NWG);

  // ---- decoder: T steps ---------------------------------------------------
  for (int t = 0; t < Tt; ++t) {
    gru_step<false>(nullptr, 0, ybuf, hcur, hnxt, hf32, bi, bh, smem, rb, cb, wave, lane);
    grid_sync(cnt, (++ep) * NWG);
    __bf16* tmp = hcur; hcur = hnxt; hnxt = tmp;
    if (cb == 0) linear_step(hcur, ybuf, out, t, bl, smem, rb, wave, lane);
    grid_sync(cnt, (++ep) * NWG);
  }
}

// ---------------------------------------------------------------------------
extern "C" void kernel_launch(void* const* d_in, const int* in_sizes, int n_in,
                              void* d_out, int out_size, void* d_ws, size_t ws_size,
                              hipStream_t stream) {
  const float* x  = (const float*)d_in[0];
  const float* Wi = (const float*)d_in[1];
  const float* Wh = (const float*)d_in[2];
  const float* bi = (const float*)d_in[3];
  const float* bh = (const float*)d_in[4];
  const float* Wl = (const float*)d_in[5];
  const float* bl = (const float*)d_in[6];
  float* out = (float*)d_out;

  // workspace layout
  char* ws = (char*)d_ws;
  size_t off = 0;
  unsigned* cnt = (unsigned*)(ws + off); off += 256;                 // barrier
  float*  hf32 = (float*)(ws + off);     off += (size_t)Bb * Hh * 4; // f32 h
  __bf16* hb0  = (__bf16*)(ws + off);    off += (size_t)Bb * Hh * 2; // bf16 h A
  __bf16* hb1  = (__bf16*)(ws + off);    off += (size_t)Bb * Hh * 2; // bf16 h B
  __bf16* ybuf = (__bf16*)(ws + off);    off += (size_t)Bb * Ii * 2; // dec input

  // Deterministic init each call (graph-capturable): zero barrier counter,
  // f32 h, and the first bf16 h buffer (h0 = 0). hb1/ybuf are write-first.
  (void)hipMemsetAsync(d_ws, 0, 256 + (size_t)Bb * Hh * 4 + (size_t)Bb * Hh * 2,
                       stream);

  (void)hipFuncSetAttribute((const void*)gru_seq2seq_persistent,
                            hipFuncAttributeMaxDynamicSharedMemorySize,
                            (int)LDS_TOTAL);

  gru_seq2seq_persistent<<<NWG, 256, LDS_TOTAL, stream>>>(
      x, Wi, Wh, bi, bh, Wl, bl, out, cnt, hf32, hb0, hb1, ybuf);
}
